// ISSF_13039520710980
// MI455X (gfx1250) — matrix operand order repaired and checked by reading, not compiled, not run on hardware
//
#include <hip/hip_runtime.h>
#include <hip/hip_bf16.h>
#include <math.h>

typedef __attribute__((ext_vector_type(16))) _Float16 v16h;
typedef __attribute__((ext_vector_type(8)))  float    v8f;

#define DEV static __device__ __forceinline__

namespace issf {

constexpr int Nb   = 4;
constexpr int T    = 2048;
constexpr int D    = 2048;   // IN_F == OUT_F
constexpr int KS   = 1024;   // K_SAL == T - K_SAL
constexpr int RD   = 128;    // OUT_F / RED
constexpr int C21  = 21;     // CLS + 1
constexpr float SCALE = 20.0f;
constexpr float LAMB  = 0.25f;

DEV float sigmoidf_(float v) { return 1.0f / (1.0f + expf(-v)); }

// ---------------------------------------------------------------------------
// WMMA fragments, layouts per CDNA5 ISA 7.12.2 (16-bit A 16x32, B 32x16).
// A is row-major MxK f16; Bt is N-major (NxK) f16 so both are contiguous loads.
// ---------------------------------------------------------------------------
DEV v16h load_a_frag(const _Float16* __restrict__ A, int lda, int m0, int k0, int lane) {
  const int row = m0 + (lane & 15);
  const int hs  = (lane >> 4) * 8;           // lanes 0-15: K 0..7 / 16..23; lanes 16-31: +8
  const _Float16* p = A + (long long)row * lda + k0 + hs;
  v16h r;
#pragma unroll
  for (int i = 0; i < 8; ++i) r[i] = p[i];
#pragma unroll
  for (int i = 0; i < 8; ++i) r[8 + i] = p[16 + i];
  return r;
}

DEV v16h load_b_frag(const _Float16* __restrict__ Bt, int ldb, int n0, int k0, int lane) {
  const int col = n0 + (lane & 15);
  const int kb  = (lane >> 4) * 8;           // lanes 0-15: K 0..7 / 16..23; lanes 16-31: +8
  const _Float16* p = Bt + (long long)col * ldb + k0 + kb;
  v16h r;
#pragma unroll
  for (int i = 0; i < 8; ++i) r[i] = p[i];
#pragma unroll
  for (int i = 0; i < 8; ++i) r[8 + i] = p[16 + i];
  return r;
}

// C = A(MxK) * Bt(NxK)^T, fp32 accumulate via v_wmma_f32_16x16x32_f16.
// 256 threads = 8 waves laid out 2(M) x 4(N); each wave owns a 32x64 tile
// (2 A-frags x 4 B-frags -> 8 accumulators), so 6 fragment loads feed 8 WMMAs.
// K is consumed 64 at a time with two fragment sets so the second set's loads
// overlap the first set's matrix ops instead of serializing on s_wait_loadcnt.
template <bool BIAS, bool RELU, bool ST32, bool ST16>
__global__ __launch_bounds__(256)
void k_gemm_wmma(const _Float16* __restrict__ A, const _Float16* __restrict__ Bt,
                 const float* __restrict__ bias,
                 float* __restrict__ C, _Float16* __restrict__ Ch,
                 int M, int Nn, int K, int lda, int ldb, int ldc,
                 long long sA, long long sB, long long sC)
{
  const int lane = threadIdx.x & 31;
  const int wave = threadIdx.x >> 5;
  const int wn   = wave & 3;
  const int wm   = wave >> 2;
  const int n0 = (blockIdx.x * 4 + wn) * 64;
  const int m0 = (blockIdx.y * 2 + wm) * 32;
  if (n0 >= Nn || m0 >= M) return;          // wave-uniform: EXEC stays all-ones for WMMA
  const long long zb = blockIdx.z;
  A  += zb * sA;
  Bt += zb * sB;

  v8f acc[2][4] = {};
  for (int k0 = 0; k0 < K; k0 += 64) {
    v16h aA[2], bA[4], aB[2], bB[4];
#pragma unroll
    for (int i = 0; i < 2; ++i) aA[i] = load_a_frag(A, lda, m0 + 16 * i, k0, lane);
#pragma unroll
    for (int j = 0; j < 4; ++j) bA[j] = load_b_frag(Bt, ldb, n0 + 16 * j, k0, lane);
#pragma unroll
    for (int i = 0; i < 2; ++i) aB[i] = load_a_frag(A, lda, m0 + 16 * i, k0 + 32, lane);
#pragma unroll
    for (int j = 0; j < 4; ++j) bB[j] = load_b_frag(Bt, ldb, n0 + 16 * j, k0 + 32, lane);
    __builtin_prefetch(A + (long long)(m0 + (lane & 15)) * lda + k0 + 256, 0, 1);
    __builtin_prefetch(Bt + (long long)(n0 + (lane & 15)) * ldb + k0 + 256, 0, 1);
#pragma unroll
    for (int i = 0; i < 2; ++i)
#pragma unroll
      for (int j = 0; j < 4; ++j)
        acc[i][j] = __builtin_amdgcn_wmma_f32_16x16x32_f16(
            false, aA[i], false, bA[j], (short)0, acc[i][j], false, false);
#pragma unroll
    for (int i = 0; i < 2; ++i)
#pragma unroll
      for (int j = 0; j < 4; ++j)
        acc[i][j] = __builtin_amdgcn_wmma_f32_16x16x32_f16(
            false, aB[i], false, bB[j], (short)0, acc[i][j], false, false);
  }

  const int cl = lane & 15;
  const int rb = (lane >> 4) * 8;
#pragma unroll
  for (int j = 0; j < 4; ++j) {
    const int col  = n0 + 16 * j + cl;
    const float bv = BIAS ? bias[col] : 0.0f;
#pragma unroll
    for (int i = 0; i < 2; ++i) {
      const int rbase = m0 + 16 * i + rb;
#pragma unroll
      for (int r = 0; r < 8; ++r) {
        float v = acc[i][j][r] + bv;
        if (RELU) v = fmaxf(v, 0.0f);
        const long long off = zb * sC + (long long)(rbase + r) * ldc + col;
        if (ST32) C[off]  = v;
        if (ST16) Ch[off] = (_Float16)v;
      }
    }
  }
}

// ---------------------------------------------------------------------------
// Elementwise / data-movement helpers
// ---------------------------------------------------------------------------
__global__ void k_f32_to_f16(const float* __restrict__ in, _Float16* __restrict__ out,
                             long long n) {
  long long i  = (long long)blockIdx.x * blockDim.x + threadIdx.x;
  long long st = (long long)gridDim.x * blockDim.x;
  for (; i < n; i += st) out[i] = (_Float16)in[i];
}

// in (R x C) f32 -> out (C x R) f16, via LDS tile
__global__ __launch_bounds__(256)
void k_transpose_f16(const float* __restrict__ in, _Float16* __restrict__ out,
                     int R, int Cc) {
  __shared__ float tile[32][33];
  const int c0 = blockIdx.x * 32, r0 = blockIdx.y * 32;
  for (int i = threadIdx.y; i < 32; i += 8) {
    int r = r0 + i, c = c0 + threadIdx.x;
    tile[i][threadIdx.x] = (r < R && c < Cc) ? in[(long long)r * Cc + c] : 0.0f;
  }
  __syncthreads();
  for (int i = threadIdx.y; i < 32; i += 8) {
    int c = c0 + i, r = r0 + threadIdx.x;
    if (c < Cc && r < R) out[(long long)c * R + r] = (_Float16)tile[threadIdx.x][i];
  }
}

// f16 batched transpose: in (R x C) -> out (C x R)
__global__ __launch_bounds__(256)
void k_transpose_h(const _Float16* __restrict__ in, _Float16* __restrict__ out,
                   int R, int Cc, long long sIn, long long sOut) {
  __shared__ _Float16 tile[32][33];
  in  += (long long)blockIdx.z * sIn;
  out += (long long)blockIdx.z * sOut;
  const int c0 = blockIdx.x * 32, r0 = blockIdx.y * 32;
  for (int i = threadIdx.y; i < 32; i += 8) {
    int r = r0 + i, c = c0 + threadIdx.x;
    tile[i][threadIdx.x] = (r < R && c < Cc) ? in[(long long)r * Cc + c] : (_Float16)0.0f;
  }
  __syncthreads();
  for (int i = threadIdx.y; i < 32; i += 8) {
    int c = c0 + i, r = r0 + threadIdx.x;
    if (c < Cc && r < R) out[(long long)c * R + r] = tile[threadIdx.x][i];
  }
}

// ---------------------------------------------------------------------------
// Norms / softmax / reductions
// ---------------------------------------------------------------------------
__global__ __launch_bounds__(256)
void k_row_invnorm(const float* __restrict__ X, float* __restrict__ inv, int Dd) {
  const long long row = blockIdx.x;
  const float* p = X + row * Dd;
  float s = 0.f;
  for (int d = threadIdx.x; d < Dd; d += 256) { float v = p[d]; s += v * v; }
  __shared__ float red[256];
  red[threadIdx.x] = s; __syncthreads();
  for (int o = 128; o > 0; o >>= 1) {
    if (threadIdx.x < o) red[threadIdx.x] += red[threadIdx.x + o];
    __syncthreads();
  }
  if (threadIdx.x == 0) inv[row] = 1.0f / (sqrtf(red[0]) + 1e-9f);
}

__global__ __launch_bounds__(256)
void k_l1n_rows(const float* __restrict__ in, float* __restrict__ out, int Dd) {
  const long long row = blockIdx.x;
  const float* p = in + row * Dd;
  float s = 0.f;
  for (int d = threadIdx.x; d < Dd; d += 256) { float v = p[d]; s += v * v; }
  __shared__ float red[256];
  __shared__ float invs;
  red[threadIdx.x] = s; __syncthreads();
  for (int o = 128; o > 0; o >>= 1) {
    if (threadIdx.x < o) red[threadIdx.x] += red[threadIdx.x + o];
    __syncthreads();
  }
  if (threadIdx.x == 0) invs = 1.0f / (sqrtf(red[0]) + 1e-9f);
  __syncthreads();
  for (int d = threadIdx.x; d < Dd; d += 256) out[row * Dd + d] = p[d] * invs;
}

// per (n,t): frm[c] = SCALE * <l1n(row), nac[c]> for 21 classes; att = sigmoid(fb)
__global__ __launch_bounds__(256)
void k_frm(const float* __restrict__ X, const float* __restrict__ invn,
           const float* __restrict__ nac, const float* __restrict__ nfg,
           float* __restrict__ frm, float* __restrict__ att) {
  const long long nt = (long long)blockIdx.y * gridDim.x + blockIdx.x;
  const float* row = X + nt * D;
  __shared__ float rowsh[D];
  __shared__ float red[256];
  for (int d = threadIdx.x; d < D; d += 256) rowsh[d] = row[d];
  __syncthreads();
  const float iv = invn[nt] * SCALE;
  for (int c = 0; c < C21 + 1; ++c) {
    const float* w = (c < C21) ? (nac + (long long)c * D) : nfg;
    float s = 0.f;
    for (int d = threadIdx.x; d < D; d += 256) s += rowsh[d] * w[d];
    red[threadIdx.x] = s; __syncthreads();
    for (int o = 128; o > 0; o >>= 1) {
      if (threadIdx.x < o) red[threadIdx.x] += red[threadIdx.x + o];
      __syncthreads();
    }
    if (threadIdx.x == 0) {
      const float v = red[0] * iv;
      if (c < C21) frm[nt * C21 + c] = v;
      else         att[nt] = sigmoidf_(v);
    }
    __syncthreads();
  }
}

// deterministic top-K selection: stable descending rank, indices ascending
__global__ __launch_bounds__(256)
void k_select(const float* __restrict__ att, int* __restrict__ sal, int* __restrict__ unsal) {
  __shared__ float sc[T];
  __shared__ int   rk[T];
  for (int t = threadIdx.x; t < T; t += 256) {
    float s = 0.f;
    for (int n = 0; n < Nb; ++n) s += att[(long long)n * T + t];
    sc[t] = s * (1.0f / Nb);                  // mean-subtract is order-invariant: skipped
  }
  __syncthreads();
  for (int t = threadIdx.x; t < T; t += 256) {
    const float st = sc[t];
    int r = 0;
    for (int j = 0; j < T; ++j) {
      const float sj = sc[j];
      r += (sj > st) || (sj == st && j < t);
    }
    rk[t] = r;
  }
  __syncthreads();
  for (int t = threadIdx.x; t < T; t += 256) {
    if (rk[t] < KS) {
      int pos = 0;
      for (int j = 0; j < t; ++j) pos += (rk[j] < KS);
      sal[pos] = t;
    } else {
      int pos = 0;
      for (int j = 0; j < t; ++j) pos += (rk[j] >= KS);
      unsal[pos] = t;
    }
  }
}

__global__ __launch_bounds__(256)
void k_gather(const float* __restrict__ X, const int* __restrict__ idx,
              float* __restrict__ f, _Float16* __restrict__ fh) {
  const int n = blockIdx.x / KS;
  const int k = blockIdx.x % KS;
  const float* src = X + ((long long)n * T + idx[k]) * D;
  float*    dst  = f  + ((long long)n * KS + k) * D;
  _Float16* dsth = fh + ((long long)n * KS + k) * D;
  for (int d = threadIdx.x; d < D; d += 256) {
    float v = src[d]; dst[d] = v; dsth[d] = (_Float16)v;
  }
}

__global__ __launch_bounds__(256)
void k_softmax_rows(float* __restrict__ X, _Float16* __restrict__ Xh, int L) {
  const long long row = blockIdx.x;
  float* p = X + row * L;
  __shared__ float red[256];
  __shared__ float bc;
  float m = -3.4e38f;
  for (int i = threadIdx.x; i < L; i += 256) m = fmaxf(m, p[i]);
  red[threadIdx.x] = m; __syncthreads();
  for (int o = 128; o > 0; o >>= 1) {
    if (threadIdx.x < o) red[threadIdx.x] = fmaxf(red[threadIdx.x], red[threadIdx.x + o]);
    __syncthreads();
  }
  if (threadIdx.x == 0) bc = red[0];
  __syncthreads();
  const float mx = bc;
  float s = 0.f;
  for (int i = threadIdx.x; i < L; i += 256) { float e = expf(p[i] - mx); p[i] = e; s += e; }
  __syncthreads();
  red[threadIdx.x] = s; __syncthreads();
  for (int o = 128; o > 0; o >>= 1) {
    if (threadIdx.x < o) red[threadIdx.x] += red[threadIdx.x + o];
    __syncthreads();
  }
  if (threadIdx.x == 0) bc = 1.0f / red[0];
  __syncthreads();
  const float inv = bc;
  for (int i = threadIdx.x; i < L; i += 256) {
    const float v = p[i] * inv;
    p[i] = v;
    if (Xh) Xh[row * L + i] = (_Float16)v;
  }
}

__global__ void k_apply_gate(const float* __restrict__ f, const float* __restrict__ g,
                             _Float16* __restrict__ fgh, long long n) {
  long long i  = (long long)blockIdx.x * blockDim.x + threadIdx.x;
  long long st = (long long)gridDim.x * blockDim.x;
  for (; i < n; i += st) fgh[i] = (_Float16)(f[i] * (1.0f + g[i]));
}

__global__ void k_combine(float* __restrict__ o, const float* __restrict__ u, long long n) {
  long long i  = (long long)blockIdx.x * blockDim.x + threadIdx.x;
  long long st = (long long)gridDim.x * blockDim.x;
  for (; i < n; i += st) o[i] = (1.0f - LAMB) * o[i] + LAMB * u[i];
}

__global__ __launch_bounds__(256)
void k_sum_att(const float* __restrict__ att, float* __restrict__ sums) {
  const int n = blockIdx.x;
  float s = 0.f;
  for (int t = threadIdx.x; t < T; t += 256) s += att[(long long)n * T + t];
  __shared__ float red[256];
  red[threadIdx.x] = s; __syncthreads();
  for (int o = 128; o > 0; o >>= 1) {
    if (threadIdx.x < o) red[threadIdx.x] += red[threadIdx.x + o];
    __syncthreads();
  }
  if (threadIdx.x == 0) sums[n] = red[0];
}

__global__ __launch_bounds__(256)
void k_sum_cw(const float* __restrict__ frm, float* __restrict__ sums) {
  const int n = blockIdx.x / C21, c = blockIdx.x % C21;
  float s = 0.f;
  for (int t = threadIdx.x; t < T; t += 256)
    s += sigmoidf_(frm[((long long)n * T + t) * C21 + c]);
  __shared__ float red[256];
  red[threadIdx.x] = s; __syncthreads();
  for (int o = 128; o > 0; o >>= 1) {
    if (threadIdx.x < o) red[threadIdx.x] += red[threadIdx.x + o];
    __syncthreads();
  }
  if (threadIdx.x == 0) sums[blockIdx.x] = red[0];
}

__global__ void k_ca_feat(const float* __restrict__ X, const float* __restrict__ att,
                          const float* __restrict__ sums, float* __restrict__ out) {
  const int gid = blockIdx.x * blockDim.x + threadIdx.x;   // n*D + d
  const int n = gid / D, d = gid % D;
  float s = 0.f;
  for (int t = 0; t < T; ++t)
    s += X[((long long)n * T + t) * D + d] * att[(long long)n * T + t];
  out[gid] = s / sums[n];
}

__global__ void k_cw_feat(const float* __restrict__ X, const float* __restrict__ frm,
                          const float* __restrict__ sums, float* __restrict__ out) {
  const int gid = blockIdx.x * blockDim.x + threadIdx.x;   // (n*C21+c)*D + d
  if (gid >= Nb * C21 * D) return;
  const int d = gid % D;
  const int nc = gid / D;
  const int n = nc / C21, c = nc % C21;
  float s = 0.f;
  for (int t = 0; t < T; ++t)
    s += X[((long long)n * T + t) * D + d] *
         sigmoidf_(frm[((long long)n * T + t) * C21 + c]);
  out[gid] = s / sums[nc];
}

// out[row] = softmax_c(SCALE * <l1n(F_row), nac[c]>)
__global__ __launch_bounds__(256)
void k_cls_softmax(const float* __restrict__ F, const float* __restrict__ nac,
                   float* __restrict__ out) {
  const long long row = blockIdx.x;
  const float* p = F + row * D;
  __shared__ float rowsh[D];
  __shared__ float red[256];
  __shared__ float sc[C21];
  __shared__ float invs;
  float s = 0.f;
  for (int d = threadIdx.x; d < D; d += 256) { float v = p[d]; rowsh[d] = v; s += v * v; }
  red[threadIdx.x] = s; __syncthreads();
  for (int o = 128; o > 0; o >>= 1) {
    if (threadIdx.x < o) red[threadIdx.x] += red[threadIdx.x + o];
    __syncthreads();
  }
  if (threadIdx.x == 0) invs = SCALE / (sqrtf(red[0]) + 1e-9f);
  __syncthreads();
  for (int c = 0; c < C21; ++c) {
    float dd = 0.f;
    const float* w = nac + (long long)c * D;
    for (int d = threadIdx.x; d < D; d += 256) dd += rowsh[d] * w[d];
    red[threadIdx.x] = dd; __syncthreads();
    for (int o = 128; o > 0; o >>= 1) {
      if (threadIdx.x < o) red[threadIdx.x] += red[threadIdx.x + o];
      __syncthreads();
    }
    if (threadIdx.x == 0) sc[c] = red[0] * invs;
    __syncthreads();
  }
  if (threadIdx.x == 0) {
    float m = -3.4e38f;
    for (int c = 0; c < C21; ++c) m = fmaxf(m, sc[c]);
    float su = 0.f;
    for (int c = 0; c < C21; ++c) { float e = expf(sc[c] - m); sc[c] = e; su += e; }
    const float inv = 1.0f / su;
    for (int c = 0; c < C21; ++c) out[row * C21 + c] = sc[c] * inv;
  }
}

// cw_pred: diagonal einsum('ncd,cd->nc') then softmax over c; one block per n
__global__ __launch_bounds__(256)
void k_cw_pred(const float* __restrict__ cwf, const float* __restrict__ nac,
               float* __restrict__ out) {
  const int n = blockIdx.x;
  __shared__ float red[256];
  __shared__ float sc[C21];
  __shared__ float nrm;
  for (int c = 0; c < C21; ++c) {
    const float* p = cwf + ((long long)n * C21 + c) * D;
    const float* w = nac + (long long)c * D;
    float ss = 0.f, dd = 0.f;
    for (int d = threadIdx.x; d < D; d += 256) { float v = p[d]; ss += v * v; dd += v * w[d]; }
    red[threadIdx.x] = ss; __syncthreads();
    for (int o = 128; o > 0; o >>= 1) {
      if (threadIdx.x < o) red[threadIdx.x] += red[threadIdx.x + o];
      __syncthreads();
    }
    if (threadIdx.x == 0) nrm = red[0];
    __syncthreads();
    red[threadIdx.x] = dd; __syncthreads();
    for (int o = 128; o > 0; o >>= 1) {
      if (threadIdx.x < o) red[threadIdx.x] += red[threadIdx.x + o];
      __syncthreads();
    }
    if (threadIdx.x == 0) sc[c] = red[0] * SCALE / (sqrtf(nrm) + 1e-9f);
    __syncthreads();
  }
  if (threadIdx.x == 0) {
    float m = -3.4e38f;
    for (int c = 0; c < C21; ++c) m = fmaxf(m, sc[c]);
    float su = 0.f;
    for (int c = 0; c < C21; ++c) { float e = expf(sc[c] - m); sc[c] = e; su += e; }
    const float inv = 1.0f / su;
    for (int c = 0; c < C21; ++c) out[n * C21 + c] = sc[c] * inv;
  }
}

} // namespace issf

extern "C" void kernel_launch(void* const* d_in, const int* in_sizes, int n_in,
                              void* d_out, int out_size, void* d_ws, size_t ws_size,
                              hipStream_t stream) {
  using namespace issf;
  (void)in_sizes; (void)n_in; (void)out_size; (void)ws_size;

  const float* x     = (const float*)d_in[0];
  const float* W_emb = (const float*)d_in[1];
  const float* b_emb = (const float*)d_in[2];
  const float* Wf1   = (const float*)d_in[3];
  const float* Wf2   = (const float*)d_in[4];
  const float* Wb1   = (const float*)d_in[5];
  const float* Wb2   = (const float*)d_in[6];
  const float* ac    = (const float*)d_in[7];
  const float* fgc   = (const float*)d_in[8];

  float* b_ca = (float*)d_out;
  float* b_cw = b_ca + Nb * C21;
  float* b_att = b_cw + Nb * C21;
  float* b_frm = b_att + (long long)Nb * T;
  float* s_ca = b_frm + (long long)Nb * T * C21;
  float* s_cw = s_ca + Nb * C21;
  float* s_att = s_cw + Nb * C21;
  float* s_frm = s_att + (long long)Nb * T;
  float* sal_feat = s_frm + (long long)Nb * T * C21;
  float* sal_raw  = sal_feat + (long long)Nb * T * D;
  float* sal_pred = sal_raw + (long long)Nb * KS * D;

  const size_t NTD = (size_t)Nb * T * D;
  const size_t NKD = (size_t)Nb * KS * D;
  const size_t NTK = (size_t)Nb * T * KS;

  char* wsp = (char*)d_ws;
  size_t off = 0;
  auto take = [&](size_t bytes) -> void* {
    void* p = wsp + off;
    off += (bytes + 255) & ~(size_t)255;
    return p;
  };

  // Unions (phase-disjoint lifetimes):
  _Float16* U1h = (_Float16*)take(NTD * 2);  // x f16  | unsal f (f32, NKD*4 == same bytes)
  float*    U1f = (float*)U1h;
  float*    U2  = (float*)take(NKD * 4);     // gate logits | attention scores S
  float*    X32 = (float*)take(NTD * 4);     // embedded x  | unsal aggregation
  float*    agg2 = X32;
  _Float16* WembT = (_Float16*)take((size_t)D * D * 2);
  _Float16* W1T   = (_Float16*)take((size_t)RD * D * 2);
  _Float16* W2T   = (_Float16*)take((size_t)D * RD * 2);
  _Float16* Xh    = (_Float16*)take(NTD * 2);
  _Float16* fh    = (_Float16*)take(NKD * 2);
  _Float16* h1h   = (_Float16*)take((size_t)Nb * KS * RD * 2);
  _Float16* fg_h  = (_Float16*)take(NKD * 2);
  _Float16* fgT_h = (_Float16*)take(NKD * 2);
  _Float16* Sh    = (_Float16*)take(NTK * 2);
  float* inv_b = (float*)take((size_t)Nb * T * 4);
  float* inv_s = (float*)take((size_t)Nb * T * 4);
  float* nac = (float*)take((size_t)C21 * D * 4);
  float* nfg = (float*)take((size_t)D * 4);
  int* sal_idx   = (int*)take((size_t)KS * 4);
  int* unsal_idx = (int*)take((size_t)(T - KS) * 4);
  float* sum_att = (float*)take((size_t)Nb * 4);
  float* sum_cw  = (float*)take((size_t)Nb * C21 * 4);
  float* ca_feat = (float*)take((size_t)Nb * D * 4);
  float* cw_feat = (float*)take((size_t)Nb * C21 * D * 4);

  const dim3 tb(256), tt(32, 8);

  // ---- Phase 0: precision conversions + normalized class centers ----------
  k_f32_to_f16<<<1024, tb, 0, stream>>>(x, U1h, (long long)NTD);
  k_transpose_f16<<<dim3(D / 32, D / 32), tt, 0, stream>>>(W_emb, WembT, D, D);
  k_l1n_rows<<<C21, tb, 0, stream>>>(ac, nac, D);
  k_l1n_rows<<<1, tb, 0, stream>>>(fgc, nfg, D);

  // ---- Phase 1: embedding GEMM, fused bias+relu, dual f32/f16 store -------
  k_gemm_wmma<true, true, true, true><<<dim3(D / 256, (Nb * T) / 64, 1), tb, 0, stream>>>(
      U1h, WembT, b_emb, X32, Xh, Nb * T, D, D, D, D, D, 0, 0, 0);

  // ---- Phase 2: norms, b-branch frm/att, saliency selection ---------------
  k_row_invnorm<<<Nb * T, tb, 0, stream>>>(X32, inv_b, D);
  k_frm<<<dim3(T, Nb), tb, 0, stream>>>(X32, inv_b, nac, nfg, b_frm, b_att);
  k_select<<<1, tb, 0, stream>>>(b_att, sal_idx, unsal_idx);

  // ---- Phase 3: b-branch prediction module --------------------------------
  k_sum_att<<<Nb, tb, 0, stream>>>(b_att, sum_att);
  k_sum_cw<<<Nb * C21, tb, 0, stream>>>(b_frm, sum_cw);
  k_ca_feat<<<(Nb * D) / 256, tb, 0, stream>>>(X32, b_att, sum_att, ca_feat);
  k_cw_feat<<<(Nb * C21 * D + 255) / 256, tb, 0, stream>>>(X32, b_frm, sum_cw, cw_feat);
  k_cls_softmax<<<Nb, tb, 0, stream>>>(ca_feat, nac, b_ca);
  k_cw_pred<<<Nb, tb, 0, stream>>>(cw_feat, nac, b_cw);

  // ---- Phases 4/5: saliency / non-saliency branches -----------------------
  auto branch = [&](const float* W1, const float* W2, const int* idx,
                    float* fdst, float* aggdst) {
    k_transpose_f16<<<dim3(RD / 32, D / 32), tt, 0, stream>>>(W1, W1T, D, RD);
    k_transpose_f16<<<dim3(D / 32, RD / 32), tt, 0, stream>>>(W2, W2T, RD, D);
    k_gather<<<Nb * KS, tb, 0, stream>>>(X32, idx, fdst, fh);
    // relu(f @ W1) -> f16   (N=128: waves with wn>=2 retire; guards are wave-uniform)
    k_gemm_wmma<false, true, false, true><<<dim3((RD + 255) / 256, (Nb * KS) / 64, 1), tb, 0, stream>>>(
        fh, W1T, nullptr, (float*)nullptr, h1h, Nb * KS, RD, D, D, D, RD, 0, 0, 0);
    // h1 @ W2 -> gate logits (f32)
    k_gemm_wmma<false, false, true, false><<<dim3(D / 256, (Nb * KS) / 64, 1), tb, 0, stream>>>(
        h1h, W2T, nullptr, U2, (_Float16*)nullptr, Nb * KS, D, RD, RD, RD, D, 0, 0, 0);
    k_softmax_rows<<<Nb * KS, tb, 0, stream>>>(U2, (_Float16*)nullptr, D);
    k_apply_gate<<<1024, tb, 0, stream>>>(fdst, U2, fg_h, (long long)NKD);
    k_transpose_h<<<dim3(D / 32, KS / 32, Nb), tt, 0, stream>>>(
        fg_h, fgT_h, KS, D, (long long)KS * D, (long long)KS * D);
    // scores: per n, (T x KS) = Xh @ fg_h^T
    k_gemm_wmma<false, false, true, false><<<dim3(KS / 256, T / 64, Nb), tb, 0, stream>>>(
        Xh, fg_h, nullptr, U2, (_Float16*)nullptr, T, KS, D, D, D, KS,
        (long long)T * D, (long long)KS * D, (long long)T * KS);
    k_softmax_rows<<<Nb * T, tb, 0, stream>>>(U2, Sh, KS);
    // aggregation: per n, (T x D) = Sh @ fgT_h^T
    k_gemm_wmma<false, false, true, false><<<dim3(D / 256, T / 64, Nb), tb, 0, stream>>>(
        Sh, fgT_h, nullptr, aggdst, (_Float16*)nullptr, T, D, KS, KS, KS, D,
        (long long)T * KS, (long long)KS * D, (long long)T * D);
  };
  branch(Wf1, Wf2, sal_idx, sal_raw, sal_feat);     // sal: f -> saliency_raw, agg -> sal_feat
  branch(Wb1, Wb2, unsal_idx, U1f, agg2);           // unsal: reuses X32 region for agg
  k_combine<<<1024, tb, 0, stream>>>(sal_feat, agg2, (long long)NTD);

  // ---- Phase 6: s-branch prediction module --------------------------------
  k_row_invnorm<<<Nb * T, tb, 0, stream>>>(sal_feat, inv_s, D);
  k_frm<<<dim3(T, Nb), tb, 0, stream>>>(sal_feat, inv_s, nac, nfg, s_frm, s_att);
  k_sum_att<<<Nb, tb, 0, stream>>>(s_att, sum_att);
  k_sum_cw<<<Nb * C21, tb, 0, stream>>>(s_frm, sum_cw);
  k_ca_feat<<<(Nb * D) / 256, tb, 0, stream>>>(sal_feat, s_att, sum_att, ca_feat);
  k_cw_feat<<<(Nb * C21 * D + 255) / 256, tb, 0, stream>>>(sal_feat, s_frm, sum_cw, cw_feat);
  k_cls_softmax<<<Nb, tb, 0, stream>>>(ca_feat, nac, s_ca);
  k_cw_pred<<<Nb, tb, 0, stream>>>(cw_feat, nac, s_cw);

  // ---- Phase 7: saliency_pred ---------------------------------------------
  k_cls_softmax<<<Nb * KS, tb, 0, stream>>>(sal_raw, nac, sal_pred);
}
